// MambaRecursiveStateAdjustmentV1_41944650613240
// MI455X (gfx1250) — matrix operand order repaired and checked by reading, compiled, not run on hardware
//
#include <hip/hip_runtime.h>
#include <math.h>

// Problem constants (from reference)
#define Cc   64
#define DI   128
#define Ssz  16
#define Rr   4
#define Kk   4
#define Hh   48
#define Ww   48
#define Ll   (Hh * Ww)      // 2304
#define Tt   4
#define RS2  36             // R + 2*S

typedef __attribute__((ext_vector_type(16))) _Float16 v16h;
typedef __attribute__((ext_vector_type(8)))  float    v8f;

// Scan-direction index mapping: xs[k][d][l] = xc_act[d][dirmap(k,l)]
__device__ __forceinline__ int dirmap(int k, int l) {
  int ll = (k >= 2) ? (Ll - 1 - l) : l;
  if (k & 1) ll = (ll % Ww) * Ww + (ll / Ww);   // H==W==48 transpose
  return ll;
}

// ---- WMMA fragment builders (gfx1250 wave32 layouts, 16x16x32 f16) ----
// A 16x32 (MxK) row-major. Per-lane data = two contiguous 8-float K-runs:
//   halves 0..7  -> K = j + (lane<16 ? 0 : 8)
//   halves 8..15 -> K = (j-8) + 16 + (lane<16 ? 0 : 8)
__device__ __forceinline__ v16h load_A_frag_nc(const float* __restrict__ row, int lane) {
  int off0 = (lane < 16) ? 0 : 8;
  float4 p0 = *(const float4*)(row + off0);
  float4 p1 = *(const float4*)(row + off0 + 4);
  float4 p2 = *(const float4*)(row + off0 + 16);
  float4 p3 = *(const float4*)(row + off0 + 20);
  v16h a;
  a[0]  = (_Float16)p0.x;  a[1]  = (_Float16)p0.y;
  a[2]  = (_Float16)p0.z;  a[3]  = (_Float16)p0.w;
  a[4]  = (_Float16)p1.x;  a[5]  = (_Float16)p1.y;
  a[6]  = (_Float16)p1.z;  a[7]  = (_Float16)p1.w;
  a[8]  = (_Float16)p2.x;  a[9]  = (_Float16)p2.y;
  a[10] = (_Float16)p2.z;  a[11] = (_Float16)p2.w;
  a[12] = (_Float16)p3.x;  a[13] = (_Float16)p3.y;
  a[14] = (_Float16)p3.z;  a[15] = (_Float16)p3.w;
  return a;
}

// Clamped variant for ragged M (x-proj, M=36): loads always safe, rows >= Mmax zeroed.
__device__ __forceinline__ v16h load_A_frag_clamp(const float* __restrict__ A, int lda,
                                                  int m0, int k0, int lane, int Mmax) {
  int m = m0 + (lane & 15);
  float scale = (m < Mmax) ? 1.0f : 0.0f;
  int mc = (m < Mmax) ? m : (Mmax - 1);
  const float* row = A + (size_t)mc * lda + k0;
  int off0 = (lane < 16) ? 0 : 8;
  float4 p0 = *(const float4*)(row + off0);
  float4 p1 = *(const float4*)(row + off0 + 4);
  float4 p2 = *(const float4*)(row + off0 + 16);
  float4 p3 = *(const float4*)(row + off0 + 20);
  v16h a;
  a[0]  = (_Float16)(p0.x * scale);  a[1]  = (_Float16)(p0.y * scale);
  a[2]  = (_Float16)(p0.z * scale);  a[3]  = (_Float16)(p0.w * scale);
  a[4]  = (_Float16)(p1.x * scale);  a[5]  = (_Float16)(p1.y * scale);
  a[6]  = (_Float16)(p1.z * scale);  a[7]  = (_Float16)(p1.w * scale);
  a[8]  = (_Float16)(p2.x * scale);  a[9]  = (_Float16)(p2.y * scale);
  a[10] = (_Float16)(p2.z * scale);  a[11] = (_Float16)(p2.w * scale);
  a[12] = (_Float16)(p3.x * scale);  a[13] = (_Float16)(p3.y * scale);
  a[14] = (_Float16)(p3.z * scale);  a[15] = (_Float16)(p3.w * scale);
  return a;
}

// B 32x16 (KxN), row-major ldb=Ll: half j -> K = k0 + (lane<16?0:8) + j + (j>=8?8:0).
// Each half-wave reads a contiguous 64B row segment -> coalesced.
__device__ __forceinline__ v16h load_B_frag(const float* __restrict__ B,
                                            int k0, int n0, int lane) {
  v16h b;
  int n  = n0 + (lane & 15);
  int kb = k0 + ((lane < 16) ? 0 : 8);
  const float* col = B + (size_t)kb * Ll + n;
  // speculative prefetch of next K-tile (global_prefetch_b8; dropped if unmapped)
  __builtin_prefetch(col + (size_t)32 * Ll, 0, 0);
#pragma unroll
  for (int j = 0; j < 16; ++j) b[j] = (_Float16)col[(size_t)(j + (j & 8)) * Ll];
  return b;
}

// ---- Generic f16-in / f32-acc WMMA GEMM (M,K compile-time; N = Ll) ----
// out[MxLl] = A[MxK] * B[KxLl] + bias[M]; one wave per 16x16 tile; no guards.
template <int M, int K>
__global__ void wmma_gemm_kernel(const float* __restrict__ A, const float* __restrict__ B,
                                 const float* __restrict__ bias, float* __restrict__ Cout) {
  int lane = threadIdx.x;
  int n0 = blockIdx.x * 16;
  int m0 = blockIdx.y * 16;
  const float* arow = A + (size_t)(m0 + (lane & 15)) * K;
  v8f acc = {};
#pragma unroll
  for (int k0 = 0; k0 < K; k0 += 32) {
    v16h a = load_A_frag_nc(arow + k0, lane);
    v16h b = load_B_frag(B, k0, n0, lane);
    acc = __builtin_amdgcn_wmma_f32_16x16x32_f16(false, a, false, b, (short)0, acc,
                                                 false, false);
  }
  int n = n0 + (lane & 15);
  int mbase = m0 + ((lane < 16) ? 0 : 8);   // multiple of 8 -> 32B-aligned bias
  float4 b0 = *(const float4*)(bias + mbase);
  float4 b1 = *(const float4*)(bias + mbase + 4);
  float* outp = Cout + (size_t)mbase * Ll + n;
  outp[0 * Ll] = acc[0] + b0.x;  outp[1 * Ll] = acc[1] + b0.y;
  outp[2 * Ll] = acc[2] + b0.z;  outp[3 * Ll] = acc[3] + b0.w;
  outp[4 * Ll] = acc[4] + b1.x;  outp[5 * Ll] = acc[5] + b1.y;
  outp[6 * Ll] = acc[6] + b1.z;  outp[7 * Ll] = acc[7] + b1.w;
}

// ---- x-proj WMMA GEMM with direction-gathered B: xdbl[k][36][L] = xp_w[k] @ xs[k] ----
__global__ void wmma_xproj_kernel(const float* __restrict__ xp_w,
                                  const float* __restrict__ xc,
                                  float* __restrict__ xdbl) {
  int lane = threadIdx.x;
  int n0 = blockIdx.x * 16;
  int m0 = blockIdx.y * 16;
  int k  = blockIdx.z;                  // direction
  const float* A = xp_w + (size_t)k * RS2 * DI;
  int n   = n0 + (lane & 15);
  int src = dirmap(k, n);               // gather column of xs from xc_act
  int kb  = (lane < 16) ? 0 : 8;
  v8f acc = {};
#pragma unroll
  for (int k0 = 0; k0 < DI; k0 += 32) {
    v16h a = load_A_frag_clamp(A, DI, m0, k0, lane, RS2);
    const float* col = xc + (size_t)(k0 + kb) * Ll + src;
    __builtin_prefetch(col + (size_t)32 * Ll, 0, 0);
    v16h b;
#pragma unroll
    for (int j = 0; j < 16; ++j) b[j] = (_Float16)col[(size_t)(j + (j & 8)) * Ll];
    acc = __builtin_amdgcn_wmma_f32_16x16x32_f16(false, a, false, b, (short)0, acc,
                                                 false, false);
  }
  int mbase = m0 + ((lane < 16) ? 0 : 8);
  float* outp = xdbl + (size_t)(k * RS2 + mbase) * Ll + n;
#pragma unroll
  for (int r = 0; r < 8; ++r) {
    if (mbase + r < RS2) outp[(size_t)r * Ll] = acc[r];
  }
}

// ---- Channel layernorm over C channels (block = C threads, one pixel per block) ----
__global__ void ln_kernel(const float* __restrict__ x, const float* __restrict__ w,
                          const float* __restrict__ b, float* __restrict__ out, int Cn) {
  __shared__ float red[128];
  int p = blockIdx.x, c = threadIdx.x;
  float v = x[(size_t)c * Ll + p];
  red[c] = v; __syncthreads();
  for (int st = Cn >> 1; st > 0; st >>= 1) { if (c < st) red[c] += red[c + st]; __syncthreads(); }
  float mu = red[0] / (float)Cn; __syncthreads();
  float d = v - mu;
  red[c] = d * d; __syncthreads();
  for (int st = Cn >> 1; st > 0; st >>= 1) { if (c < st) red[c] += red[c + st]; __syncthreads(); }
  float var = red[0] / (float)Cn;
  out[(size_t)c * Ll + p] = d * rsqrtf(var + 1e-6f) * w[c] + b[c];
}

// cur = prev * exp(sigma)
__global__ void curinit_kernel(const float* __restrict__ prev, const float* __restrict__ sig,
                               float* __restrict__ cur, int n) {
  int i = blockIdx.x * blockDim.x + threadIdx.x;
  if (i < n) cur[i] = prev[i] * expf(sig[i]);
}

// kl = exp(lp)*(lp-lq), log-softmax over C=64 channels per pixel
__global__ void kl_kernel(const float* __restrict__ cur, const float* __restrict__ prev,
                          float* __restrict__ out) {
  __shared__ float red[Cc];
  int p = blockIdx.x, c = threadIdx.x;
  float pv = prev[(size_t)c * Ll + p];
  float qv = cur[(size_t)c * Ll + p];
  red[c] = pv; __syncthreads();
  for (int st = Cc >> 1; st > 0; st >>= 1) { if (c < st) red[c] = fmaxf(red[c], red[c + st]); __syncthreads(); }
  float mp = red[0]; __syncthreads();
  red[c] = expf(pv - mp); __syncthreads();
  for (int st = Cc >> 1; st > 0; st >>= 1) { if (c < st) red[c] += red[c + st]; __syncthreads(); }
  float lsep = mp + logf(red[0]); __syncthreads();
  red[c] = qv; __syncthreads();
  for (int st = Cc >> 1; st > 0; st >>= 1) { if (c < st) red[c] = fmaxf(red[c], red[c + st]); __syncthreads(); }
  float mq = red[0]; __syncthreads();
  red[c] = expf(qv - mq); __syncthreads();
  for (int st = Cc >> 1; st > 0; st >>= 1) { if (c < st) red[c] += red[c + st]; __syncthreads(); }
  float lseq = mq + logf(red[0]);
  float lp = pv - lsep, lq = qv - lseq;
  out[(size_t)c * Ll + p] = expf(lp) * (lp - lq);
}

// depthwise 3x3 SAME conv on rows [0,DI) of xz, + bias, SiLU
__global__ void conv_silu_kernel(const float* __restrict__ xz, const float* __restrict__ cw,
                                 const float* __restrict__ cb, float* __restrict__ xca) {
  int idx = blockIdx.x * blockDim.x + threadIdx.x;
  if (idx >= DI * Ll) return;
  int p = idx % Ll, d = idx / Ll;
  int h = p / Ww, w = p % Ww;
  float acc = cb[d];
  const float* wd = cw + d * 9;
  const float* xd = xz + (size_t)d * Ll;
#pragma unroll
  for (int i = 0; i < 3; ++i) {
    int h2 = h + i - 1;
    if (h2 < 0 || h2 >= Hh) continue;
#pragma unroll
    for (int j = 0; j < 3; ++j) {
      int w2 = w + j - 1;
      if (w2 < 0 || w2 >= Ww) continue;
      acc += wd[i * 3 + j] * xd[h2 * Ww + w2];
    }
  }
  xca[idx] = acc / (1.0f + expf(-acc));       // SiLU
}

// dt[k][d][l] = softplus(sum_r dt_w[k][d][r]*xdbl[k][r][l] + dt_b[k][d])   (K=4 -> VALU)
__global__ void dt_kernel(const float* __restrict__ dtw, const float* __restrict__ dtb,
                          const float* __restrict__ xdbl, float* __restrict__ dt) {
  int idx = blockIdx.x * blockDim.x + threadIdx.x;
  if (idx >= Kk * DI * Ll) return;
  int l = idx % Ll;
  int kd = idx / Ll;
  int d = kd % DI, k = kd / DI;
  float acc = dtb[k * DI + d];
#pragma unroll
  for (int r = 0; r < Rr; ++r)
    acc += dtw[(size_t)(k * DI + d) * Rr + r] * xdbl[(size_t)(k * RS2 + r) * Ll + l];
  dt[idx] = (acc > 20.0f) ? acc : log1pf(expf(acc));
}

// Selective scan: S=16 states mapped across half-wave lanes; serial in L.
// xs index walked incrementally (no div/mod in the hot loop).
__global__ void scan_kernel(const float* __restrict__ dt, const float* __restrict__ xc,
                            const float* __restrict__ xdbl, const float* __restrict__ Alog,
                            const float* __restrict__ Ds, float* __restrict__ y) {
  int tid = blockIdx.x * blockDim.x + threadIdx.x;
  int s = tid & 15;
  int grp = tid >> 4;                   // (k,d) group, 0..511
  if (grp >= Kk * DI) return;
  int d = grp % DI, k = grp / DI;
  float As = -expf(Alog[(size_t)(k * DI + d) * Ssz + s]);
  float Dv = Ds[k * DI + d];
  const float* dtp = dt   + (size_t)(k * DI + d) * Ll;
  const float* Bp  = xdbl + (size_t)(k * RS2 + Rr + s) * Ll;
  const float* Cp  = xdbl + (size_t)(k * RS2 + Rr + Ssz + s) * Ll;
  const float* xd  = xc   + (size_t)d * Ll;
  float* yp = y + (size_t)(k * DI + d) * Ll;
  // incremental xs index: dir0:+1 from 0; dir1:+48 (wrap -2303); dir2:-1 from L-1; dir3:-48 (wrap +2303)
  int idx  = (k < 2) ? ((k == 0) ? 0 : 0) : Ll - 1;
  int step = (k == 0) ? 1 : (k == 1) ? Ww : (k == 2) ? -1 : -Ww;
  int wrap = (k == 1) ? -(Ll - 1) : (k == 3) ? (Ll - 1) : 0;
  int cnt = 0;
  float h = 0.0f;
  for (int l = 0; l < Ll; ++l) {
    float dtv = dtp[l];
    float xv  = xd[idx];
    float dA  = __expf(dtv * As);
    h = h * dA + (dtv * xv) * Bp[l];
    float c = h * Cp[l];
    c += __shfl_xor(c, 8, 16);
    c += __shfl_xor(c, 4, 16);
    c += __shfl_xor(c, 2, 16);
    c += __shfl_xor(c, 1, 16);
    if (s == 0) yp[l] = c + Dv * xv;
    idx += step;
    if (++cnt == Ww) { cnt = 0; idx += wrap; }   // no-op (wrap=0) for dirs 0,2
  }
}

// merge 4 directions + LN over DI + SiLU(z) gate
__global__ void merge_kernel(const float* __restrict__ y, const float* __restrict__ xz,
                             const float* __restrict__ onw, const float* __restrict__ onb,
                             float* __restrict__ yy) {
  __shared__ float red[DI];
  int p = blockIdx.x, d = threadIdx.x;
  int h = p / Ww, w = p % Ww;
  int tp = w * Ww + h;
  float v = y[(size_t)(0 * DI + d) * Ll + p]
          + y[(size_t)(1 * DI + d) * Ll + tp]
          + y[(size_t)(2 * DI + d) * Ll + (Ll - 1 - p)]
          + y[(size_t)(3 * DI + d) * Ll + (Ll - 1 - tp)];
  red[d] = v; __syncthreads();
  for (int st = DI >> 1; st > 0; st >>= 1) { if (d < st) red[d] += red[d + st]; __syncthreads(); }
  float mu = red[0] / (float)DI; __syncthreads();
  float dv = v - mu;
  red[d] = dv * dv; __syncthreads();
  for (int st = DI >> 1; st > 0; st >>= 1) { if (d < st) red[d] += red[d + st]; __syncthreads(); }
  float var = red[0] / (float)DI;
  float ln = dv * rsqrtf(var + 1e-6f) * onw[d] + onb[d];
  float z = xz[(size_t)(DI + d) * Ll + p];
  yy[(size_t)d * Ll + p] = ln * (z / (1.0f + expf(-z)));
}

// next = cur*exp(s) + bias
__global__ void update_kernel(const float* __restrict__ cur, const float* __restrict__ sb,
                              const float* __restrict__ bb, float* __restrict__ nxt, int n) {
  int i = blockIdx.x * blockDim.x + threadIdx.x;
  if (i < n) nxt[i] = cur[i] * expf(sb[i]) + bb[i];
}

__global__ void copy_kernel(const float* __restrict__ src, float* __restrict__ dst, int n) {
  int i = blockIdx.x * blockDim.x + threadIdx.x;
  if (i < n) dst[i] = src[i];
}

// ---- Host-side: one full _ss2d pipeline ----
static void run_ss2d(const float* xin, int br,
                     const float* g_inw, const float* g_inb,
                     const float* g_cw, const float* g_cb,
                     const float* g_xpw, const float* g_dtw, const float* g_dtb,
                     const float* g_alog, const float* g_ds,
                     const float* g_onw, const float* g_onb,
                     const float* g_ow, const float* g_ob,
                     float* xz, float* xca, float* xdbl, float* dt, float* ybuf, float* yy,
                     float* target, hipStream_t stream) {
  // in-proj: 256x64 @ 64xL
  wmma_gemm_kernel<2 * DI, Cc><<<dim3(Ll / 16, (2 * DI) / 16), 32, 0, stream>>>(
      g_inw + (size_t)br * 2 * DI * Cc, xin, g_inb + (size_t)br * 2 * DI, xz);
  // depthwise conv + SiLU
  conv_silu_kernel<<<(DI * Ll + 255) / 256, 256, 0, stream>>>(
      xz, g_cw + (size_t)br * DI * 9, g_cb + (size_t)br * DI, xca);
  // x-proj: 4 directions, 36x128 @ 128xL (B gathered)
  wmma_xproj_kernel<<<dim3(Ll / 16, 3, Kk), 32, 0, stream>>>(
      g_xpw + (size_t)br * Kk * RS2 * DI, xca, xdbl);
  // dt = softplus(dt_w @ dts + dt_b)
  dt_kernel<<<(Kk * DI * Ll + 255) / 256, 256, 0, stream>>>(
      g_dtw + (size_t)br * Kk * DI * Rr, g_dtb + (size_t)br * Kk * DI, xdbl, dt);
  // selective scan (512 lane-groups of 16 states)
  scan_kernel<<<(Kk * DI * Ssz + 255) / 256, 256, 0, stream>>>(
      dt, xca, xdbl, g_alog + (size_t)br * Kk * DI * Ssz, g_ds + (size_t)br * Kk * DI, ybuf);
  // merge + LN + gate
  merge_kernel<<<Ll, DI, 0, stream>>>(ybuf, xz, g_onw + (size_t)br * DI,
                                      g_onb + (size_t)br * DI, yy);
  // out-proj: 64x128 @ 128xL
  wmma_gemm_kernel<Cc, DI><<<dim3(Ll / 16, Cc / 16), 32, 0, stream>>>(
      g_ow + (size_t)br * Cc * DI, yy, g_ob + (size_t)br * Cc, target);
}

extern "C" void kernel_launch(void* const* d_in, const int* in_sizes, int n_in,
                              void* d_out, int out_size, void* d_ws, size_t ws_size,
                              hipStream_t stream) {
  (void)in_sizes; (void)n_in; (void)out_size; (void)ws_size;
  const float* g_img  = (const float*)d_in[0];
  const float* g_dz   = (const float*)d_in[1];
  const float* g_sig  = (const float*)d_in[2];
  const float* g_inw  = (const float*)d_in[3];
  const float* g_inb  = (const float*)d_in[4];
  const float* g_cw   = (const float*)d_in[5];
  const float* g_cb   = (const float*)d_in[6];
  const float* g_xpw  = (const float*)d_in[7];
  const float* g_dtw  = (const float*)d_in[8];
  const float* g_dtb  = (const float*)d_in[9];
  const float* g_alog = (const float*)d_in[10];
  const float* g_ds   = (const float*)d_in[11];
  const float* g_onw  = (const float*)d_in[12];
  const float* g_onb  = (const float*)d_in[13];
  const float* g_ow   = (const float*)d_in[14];
  const float* g_ob   = (const float*)d_in[15];
  const float* g_niw  = (const float*)d_in[16];
  const float* g_nib  = (const float*)d_in[17];
  const float* g_ndw  = (const float*)d_in[18];
  const float* g_ndb  = (const float*)d_in[19];

  float* ws = (float*)d_ws;
  const size_t CL = (size_t)Cc * Ll;          // 147456
  float* bufA = ws;
  float* bufB = bufA + CL;
  float* bufC = bufB + CL;
  float* xbuf = bufC + CL;
  float* xz   = xbuf + CL;                    // 2*DI*L
  float* xca  = xz   + (size_t)2 * DI * Ll;   // DI*L
  float* xdbl = xca  + (size_t)DI * Ll;       // K*36*L
  float* dt   = xdbl + (size_t)Kk * RS2 * Ll; // K*DI*L
  float* ybuf = dt   + (size_t)Kk * DI * Ll;  // K*DI*L
  float* yy   = ybuf + (size_t)Kk * DI * Ll;  // DI*L
  float* sbuf = yy   + (size_t)DI * Ll;       // C*L
  float* bbuf = sbuf + CL;                    // C*L

  // prev = LN(difficult_zone); cur = prev*exp(sigma)
  ln_kernel<<<Ll, Cc, 0, stream>>>(g_dz, g_ndw, g_ndb, bufA, Cc);
  curinit_kernel<<<((int)CL + 255) / 256, 256, 0, stream>>>(bufA, g_sig, bufB, (int)CL);

  float* prevp = bufA; float* curp = bufB; float* nextp = bufC;

  for (int t = 0; t < Tt; ++t) {
    // branch 0 on KL(cur, prev)
    kl_kernel<<<Ll, Cc, 0, stream>>>(curp, prevp, xbuf);
    run_ss2d(xbuf, 0, g_inw, g_inb, g_cw, g_cb, g_xpw, g_dtw, g_dtb, g_alog, g_ds,
             g_onw, g_onb, g_ow, g_ob, xz, xca, xdbl, dt, ybuf, yy, sbuf, stream);
    // branch 1 on LN(image[t])
    ln_kernel<<<Ll, Cc, 0, stream>>>(g_img + (size_t)t * CL, g_niw, g_nib, xbuf, Cc);
    run_ss2d(xbuf, 1, g_inw, g_inb, g_cw, g_cb, g_xpw, g_dtw, g_dtb, g_alog, g_ds,
             g_onw, g_onb, g_ow, g_ob, xz, xca, xdbl, dt, ybuf, yy, bbuf, stream);
    // cur, prev = cur*exp(s)+bias, cur
    update_kernel<<<((int)CL + 255) / 256, 256, 0, stream>>>(curp, sbuf, bbuf, nextp, (int)CL);
    float* tmp = prevp; prevp = curp; curp = nextp; nextp = tmp;
  }
  copy_kernel<<<((int)CL + 255) / 256, 256, 0, stream>>>(curp, (float*)d_out, (int)CL);
}
